// BiFQRNNREDC3D_14946486190560
// MI455X (gfx1250) — matrix operand order repaired and checked by reading, not compiled, not run on hardware
//
#include <hip/hip_runtime.h>
#include <cmath>

typedef __attribute__((ext_vector_type(16))) __bf16 v16bf;
typedef __attribute__((ext_vector_type(8)))  __bf16 v8bf;
typedef __attribute__((ext_vector_type(8)))  float  v8f;
typedef __attribute__((ext_vector_type(4)))  unsigned int u32x4;
typedef __attribute__((ext_vector_type(8)))  int    i32x8;
typedef __attribute__((ext_vector_type(4)))  int    i32x4;

#define TT 31
#define HH 128
#define WW 128
#define VV (TT*HH*WW)          // 507904 voxels
#define HWP (HH*WW)

#ifndef __has_builtin
#define __has_builtin(x) 0
#endif
#if defined(__AMDGCN__) && __has_builtin(__builtin_amdgcn_tensor_load_to_lds) && __has_builtin(__builtin_amdgcn_s_wait_tensorcnt)
#define HAVE_TDM 1
#else
#define HAVE_TDM 0
#endif

// ---------------------------------------------------------------------------
// Weight prep: f32 OIDHW (or deconv IODHW+flip) -> bf16 [co][28 taps][16 ci]
// ---------------------------------------------------------------------------
__global__ void prep_w_kernel(const float* __restrict__ Wsrc, __bf16* __restrict__ wb,
                              int Cout, int Cin, int deconv, int mtiles) {
    int idx = blockIdx.x * 256 + threadIdx.x;
    int total = mtiles * 16 * 448;
    if (idx >= total) return;
    int ci  = idx & 15;
    int tap = (idx >> 4) % 28;
    int co  = idx / 448;
    float v = 0.f;
    if (co < Cout && ci < Cin && tap < 27) {
        if (deconv) v = Wsrc[((size_t)ci * Cout + co) * 27 + (26 - tap)];  // swap(0,1)+flip
        else        v = Wsrc[((size_t)co * Cin  + ci) * 27 + tap];
    }
    wb[idx] = (__bf16)v;
}

// ---------------------------------------------------------------------------
// BatchNorm stats: one block per channel -> scale/shift
// ---------------------------------------------------------------------------
__global__ void bn_stats_kernel(const float* __restrict__ x, const float* __restrict__ gamma,
                                const float* __restrict__ beta, float* __restrict__ ss, int Vn) {
    __shared__ float s1[256];
    __shared__ float s2[256];
    int c = blockIdx.x;
    const float* xc = x + (size_t)c * Vn;
    float a = 0.f, b = 0.f;
    for (int i = threadIdx.x; i < Vn; i += 256) { float v = xc[i]; a += v; b += v * v; }
    s1[threadIdx.x] = a; s2[threadIdx.x] = b;
    __syncthreads();
    for (int s = 128; s > 0; s >>= 1) {
        if (threadIdx.x < (unsigned)s) {
            s1[threadIdx.x] += s1[threadIdx.x + s];
            s2[threadIdx.x] += s2[threadIdx.x + s];
        }
        __syncthreads();
    }
    if (threadIdx.x == 0) {
        float mu  = s1[0] / (float)Vn;
        float var = s2[0] / (float)Vn - mu * mu;
        float sc  = gamma[c] * rsqrtf(var + 1e-5f);
        ss[c]      = sc;
        ss[16 + c] = beta[c] - mu * sc;
    }
}

// ---------------------------------------------------------------------------
// Normalize + convert to bf16, channel-minor layout: actbf[vox*16 + ci]
// (ci >= Cin zero-padded so GEMM K-slices are uniform 16)
// ---------------------------------------------------------------------------
__global__ void bn_norm_kernel(const float* __restrict__ x, const float* __restrict__ ss,
                               __bf16* __restrict__ actbf, int Cin, int Vn) {
    int vox = blockIdx.x * 256 + threadIdx.x;
    if (vox >= Vn) return;
    __bf16 o[16];
#pragma unroll
    for (int ci = 0; ci < 16; ++ci) {
        float v = 0.f;
        if (ci < Cin) v = x[(size_t)ci * Vn + vox] * ss[ci] + ss[16 + ci];
        o[ci] = (__bf16)v;
    }
    *reinterpret_cast<v16bf*>(actbf + (size_t)vox * 16) = *reinterpret_cast<v16bf*>(o);
}

// ---------------------------------------------------------------------------
// Implicit-GEMM 3x3x3 conv via v_wmma_f32_16x16x32_bf16.
// Block = 256 thr (8 waves); block handles one (z,y) row of 128 x-positions,
// each wave one 16-voxel x-tile x up to 3 tiles of 16 output channels.
// Weights staged to LDS via TDM tensor_load_to_lds (fallback: coop loads).
// ---------------------------------------------------------------------------
__global__ __launch_bounds__(256) void conv_wmma_kernel(
    const __bf16* __restrict__ actbf, const __bf16* __restrict__ wb,
    const float* __restrict__ bias, float* __restrict__ g, int mtiles, int Cout) {
    __shared__ __align__(16) __bf16 Wlds[48 * 448];   // 43008 B, sole LDS object (offset 0)

    const int tid    = threadIdx.x;
    const int lane   = tid & 31;
    const int wid    = tid >> 5;
    const int khalf  = lane >> 4;
    const int welems = mtiles * 16 * 448;

#if HAVE_TDM
    if (wid == 0) {
        unsigned long long ga = (unsigned long long)(uintptr_t)wb;
        unsigned nelem = (unsigned)welems;            // 2-byte elements, 1-row tile
        u32x4 g0;
        g0[0] = 1u;                                   // count=1, no gather
        g0[1] = 0u;                                   // lds_addr = 0 (Wlds)
        g0[2] = (unsigned)(ga & 0xFFFFFFFFull);
        g0[3] = (unsigned)((ga >> 32) & 0x1FFFFFFull) | (2u << 30);   // type=2 (image)
        i32x8 g1;
        g1[0] = (int)(1u << 16);                      // data_size = 1 (2 bytes)
        g1[1] = (int)((nelem & 0xFFFFu) << 16);       // tensor_dim0[15:0]
        g1[2] = (int)(((nelem >> 16) & 0xFFFFu) | (1u << 16));  // dim0 hi | tensor_dim1=1
        g1[3] = (int)((nelem & 0xFFFFu) << 16);       // tile_dim0
        g1[4] = 1;                                    // tile_dim1 = 1
        g1[5] = (int)nelem;                           // tensor_dim0_stride lo
        g1[6] = 0;
        g1[7] = 0;
        i32x4 gz; gz[0] = 0; gz[1] = 0; gz[2] = 0; gz[3] = 0;
#if __clang_major__ >= 23
        i32x8 gz8; gz8[0]=0; gz8[1]=0; gz8[2]=0; gz8[3]=0; gz8[4]=0; gz8[5]=0; gz8[6]=0; gz8[7]=0;
        __builtin_amdgcn_tensor_load_to_lds(g0, g1, gz, gz, gz8, 0);
#else
        __builtin_amdgcn_tensor_load_to_lds(g0, g1, gz, gz, 0);
#endif
        __builtin_amdgcn_s_wait_tensorcnt(0);
    }
#else
    for (int i = tid * 8; i < welems; i += 256 * 8)
        *reinterpret_cast<v8bf*>(&Wlds[i]) = *reinterpret_cast<const v8bf*>(wb + i);
#endif
    __syncthreads();

    const int z  = blockIdx.x / HH;
    const int y  = blockIdx.x % HH;
    const int x0 = (wid << 4) + (lane & 15);          // this lane's output x (N index)
    const size_t rowvox = ((size_t)z * HH + y) * WW;
    const size_t ovox   = rowvox + x0;

    __builtin_prefetch(actbf + rowvox * 16, 0, 3);    // global_prefetch path

    v8f acc0 = {}; v8f acc1 = {}; v8f acc2 = {};

    for (int tp = 0; tp < 14; ++tp) {                 // 14 K=32 chunks = taps (2tp,2tp+1)
        // ---- B fragment: lanes 0-15 carry tap 2tp, lanes 16-31 tap 2tp+1 ----
        const int tap = 2 * tp + khalf;
        v16bf bfrag = {};
        if (tap < 27) {
            int dz = tap / 9 - 1, dy = (tap % 9) / 3 - 1, dx = tap % 3 - 1;
            int zz = z + dz, yy = y + dy, xx = x0 + dx;
            if ((unsigned)zz < TT && (unsigned)yy < HH && (unsigned)xx < WW) {
                size_t vox = ((size_t)zz * HH + yy) * WW + xx;
                const v8bf* bp = reinterpret_cast<const v8bf*>(actbf + vox * 16);
                ((v8bf*)&bfrag)[0] = bp[0];           // K = khalf*16 + 0..7  (ci 0..7)
                ((v8bf*)&bfrag)[1] = bp[1];           // K = khalf*16 + 8..15 (ci 8..15)
            }
        }
        // ---- A fragments from LDS (contiguous K = khalf*16 + s, same as B) + WMMA ----
        auto load_a = [&](int co) -> v16bf {
            v16bf a;
            int wo = co * 448 + tp * 32 + khalf * 16; // [co][tap=2tp+khalf][ci=0..]
            ((v8bf*)&a)[0] = *reinterpret_cast<const v8bf*>(&Wlds[wo]);
            ((v8bf*)&a)[1] = *reinterpret_cast<const v8bf*>(&Wlds[wo + 8]); // ci 8..15
            return a;
        };
        {
            v16bf a = load_a(lane & 15);
            acc0 = __builtin_amdgcn_wmma_f32_16x16x32_bf16(false, a, false, bfrag,
                                                           (short)0, acc0, false, false);
        }
        if (mtiles > 1) {
            v16bf a = load_a(16 + (lane & 15));
            acc1 = __builtin_amdgcn_wmma_f32_16x16x32_bf16(false, a, false, bfrag,
                                                           (short)0, acc1, false, false);
        }
        if (mtiles > 2) {
            v16bf a = load_a(32 + (lane & 15));
            acc2 = __builtin_amdgcn_wmma_f32_16x16x32_bf16(false, a, false, bfrag,
                                                           (short)0, acc2, false, false);
        }
    }

    // ---- store: D vgpr r holds row M = r + 8*khalf, col N = lane&15 ----
    auto store_acc = [&](const v8f& a, int m) {
#pragma unroll
        for (int r = 0; r < 8; ++r) {
            int co = m * 16 + khalf * 8 + r;
            if (co < Cout) {
                float v = a[r];
                if (bias) v += bias[co];
                g[(size_t)co * VV + ovox] = v;
            }
        }
    };
    store_acc(acc0, 0);
    if (mtiles > 1) store_acc(acc1, 1);
    if (mtiles > 2) store_acc(acc2, 2);
}

// ---------------------------------------------------------------------------
// Bidirectional QRNN scan over T: out = fmul(tanh(Z), sig(F1)) +
// fmul(tanh(Z), sig(F2), rev) (+skip). One thread per (c, h, w) column.
// ---------------------------------------------------------------------------
__global__ void scan_kernel(const float* __restrict__ g, const float* __restrict__ skip,
                            float* __restrict__ out, int Cc) {
    int idx = blockIdx.x * 256 + threadIdx.x;
    int plane = idx % HWP;
    int c = idx / HWP;
    if (c >= Cc) return;
    const float* Z  = g + (size_t)c * VV;
    const float* F1 = g + (size_t)(Cc + c) * VV;
    const float* F2 = g + (size_t)(2 * Cc + c) * VV;
    float* oc = out + (size_t)c * VV;
    const float* sc = skip ? (skip + (size_t)c * VV) : nullptr;
    float h = 0.f;
    for (int t = 0; t < TT; ++t) {
        size_t i = (size_t)t * HWP + plane;
        float zv = tanhf(Z[i]);
        float f  = 1.f / (1.f + __expf(-F1[i]));
        h = f * h + (1.f - f) * zv;
        oc[i] = h;
    }
    h = 0.f;
    for (int t = TT - 1; t >= 0; --t) {
        size_t i = (size_t)t * HWP + plane;
        float zv = tanhf(Z[i]);
        float f  = 1.f / (1.f + __expf(-F2[i]));
        h = f * h + (1.f - f) * zv;
        float v = oc[i] + h;
        if (sc) v += sc[i];
        oc[i] = v;
    }
}

// ---------------------------------------------------------------------------
// Orchestration
// ---------------------------------------------------------------------------
extern "C" void kernel_launch(void* const* d_in, const int* in_sizes, int n_in,
                              void* d_out, int out_size, void* d_ws, size_t ws_size,
                              hipStream_t stream) {
    (void)in_sizes; (void)n_in; (void)out_size; (void)ws_size;
    const float* x        = (const float*)d_in[0];
    const float* fe_gamma = (const float*)d_in[1];
    const float* fe_beta  = (const float*)d_in[2];
    const float* fe_W     = (const float*)d_in[3];
    const float* enc_gamma= (const float*)d_in[4];
    const float* enc_beta = (const float*)d_in[5];
    const float* enc_W    = (const float*)d_in[6];
    const float* dec_gamma= (const float*)d_in[7];
    const float* dec_beta = (const float*)d_in[8];
    const float* dec_W    = (const float*)d_in[9];
    const float* rec_gamma= (const float*)d_in[10];
    const float* rec_beta = (const float*)d_in[11];
    const float* rec_W    = (const float*)d_in[12];
    const float* rec_b    = (const float*)d_in[13];

    size_t off = 0;
    auto carve = [&](size_t bytes) -> char* {
        char* p = (char*)d_ws + off;
        off += (bytes + 255) & ~(size_t)255;
        return p;
    };
    __bf16* wb[10];
    for (int i = 0; i < 10; ++i) wb[i] = (__bf16*)carve(48 * 448 * sizeof(__bf16));
    float*  ss    = (float*)carve(32 * sizeof(float));
    __bf16* actbf = (__bf16*)carve((size_t)VV * 16 * sizeof(__bf16));
    float*  gbuf  = (float*)carve((size_t)48 * VV * sizeof(float));
    float*  A1 = (float*)carve((size_t)16 * VV * sizeof(float));
    float*  A2 = (float*)carve((size_t)16 * VV * sizeof(float));
    float*  A3 = (float*)carve((size_t)16 * VV * sizeof(float));
    float*  A4 = (float*)carve((size_t)16 * VV * sizeof(float));
    float*  P  = (float*)carve((size_t)16 * VV * sizeof(float));
    float*  Q  = (float*)carve((size_t)16 * VV * sizeof(float));

    // --- pack all weights to bf16 GEMM layout ---
    prep_w_kernel<<<(3 * 16 * 448 + 255) / 256, 256, 0, stream>>>(fe_W, wb[0], 48, 1, 0, 3);
    for (int i = 0; i < 4; ++i)
        prep_w_kernel<<<(3 * 16 * 448 + 255) / 256, 256, 0, stream>>>(
            enc_W + (size_t)i * 48 * 16 * 27, wb[1 + i], 48, 16, 0, 3);
    for (int i = 0; i < 4; ++i)
        prep_w_kernel<<<(3 * 16 * 448 + 255) / 256, 256, 0, stream>>>(
            dec_W + (size_t)i * 16 * 48 * 27, wb[5 + i], 48, 16, 1, 3);
    prep_w_kernel<<<(1 * 16 * 448 + 255) / 256, 256, 0, stream>>>(rec_W, wb[9], 3, 16, 1, 1);

    auto layer = [&](const float* in, int Cin, const float* gamma, const float* beta,
                     const __bf16* w, int mtiles, int Cout, const float* bias,
                     float* outb, const float* skip, int Cc) {
        bn_stats_kernel<<<Cin, 256, 0, stream>>>(in, gamma, beta, ss, VV);
        bn_norm_kernel<<<VV / 256, 256, 0, stream>>>(in, ss, actbf, Cin, VV);
        conv_wmma_kernel<<<TT * HH, 256, 0, stream>>>(actbf, w, bias, gbuf, mtiles, Cout);
        scan_kernel<<<(Cc * HWP) / 256, 256, 0, stream>>>(gbuf, skip, outb, Cc);
    };

    layer(x,  1,  fe_gamma,       fe_beta,       wb[0], 3, 48, nullptr, A1, nullptr, 16);
    layer(A1, 16, enc_gamma + 0,  enc_beta + 0,  wb[1], 3, 48, nullptr, A2, nullptr, 16);
    layer(A2, 16, enc_gamma + 16, enc_beta + 16, wb[2], 3, 48, nullptr, A3, nullptr, 16);
    layer(A3, 16, enc_gamma + 32, enc_beta + 32, wb[3], 3, 48, nullptr, A4, nullptr, 16);
    layer(A4, 16, enc_gamma + 48, enc_beta + 48, wb[4], 3, 48, nullptr, P,  nullptr, 16);
    layer(P,  16, dec_gamma + 0,  dec_beta + 0,  wb[5], 3, 48, nullptr, Q,  A4,      16);
    layer(Q,  16, dec_gamma + 16, dec_beta + 16, wb[6], 3, 48, nullptr, P,  A3,      16);
    layer(P,  16, dec_gamma + 32, dec_beta + 32, wb[7], 3, 48, nullptr, Q,  A2,      16);
    layer(Q,  16, dec_gamma + 48, dec_beta + 48, wb[8], 3, 48, nullptr, P,  A1,      16);
    layer(P,  16, rec_gamma,      rec_beta,      wb[9], 1, 3,  rec_b, (float*)d_out, x, 1);
}
